// EvoformerBlock_15564961481079
// MI455X (gfx1250) — matrix operand order repaired and checked by reading, not compiled, not run on hardware
//
#include <hip/hip_runtime.h>
#include <hip/hip_bf16.h>
#include <math.h>

typedef __attribute__((ext_vector_type(16))) __bf16 v16bf;
typedef __attribute__((ext_vector_type(8)))  __bf16 v8bf;
typedef __attribute__((ext_vector_type(8)))  float  v8f;

#define DEV __device__ __forceinline__

DEV v8bf zero8() {
  v8bf z;
#pragma unroll
  for (int i = 0; i < 8; i++) z[i] = (__bf16)0.0f;
  return z;
}

DEV v16bf mk16(v8bf lo, v8bf hi) {
  v16bf r;
#pragma unroll
  for (int i = 0; i < 8; i++) { r[i] = lo[i]; r[i + 8] = hi[i]; }
  return r;
}

// Low 32 bits of a generic pointer to __shared__ = wave-relative LDS byte offset
DEV unsigned lds_off32(const void* p) {
  return (unsigned)(unsigned long long)p;
}

// Async memory->LDS copy of 16 bytes (gfx1250 GLOBAL_LOAD_ASYNC_TO_LDS_B128).
// Tracked by ASYNCcnt; caller must s_wait_asynccnt before consuming LDS.
DEV void async_cp16(void* lds, const void* gaddr) {
  asm volatile("global_load_async_to_lds_b128 %0, %1, off"
               :: "v"(lds_off32(lds)), "v"(gaddr) : "memory");
}
DEV void wait_async0() {
  asm volatile("s_wait_asynccnt 0" ::: "memory");
}

// ---------------------------------------------------------------------------
// Tiled bf16 WMMA GEMM.  C[M,N] = act(alpha * A(MxK) * B + bias)
//   TRANSB=1: B given as [N][K] row-major (pre-transposed weights / K-matrix)
//   TRANSB=0: B given as [K][N] row-major
//   FULL=1:   M%128==0 && N%128==0 -> no guards, async-LDS staging
// Batched: per blockIdx.z, offsets = (z/bmod)*s?o + (z%bmod)*s?i
// Output element (row,col) goes to out[obase + row*orow + col*ocol]
// ---------------------------------------------------------------------------
template <int TRANSB, int ACT, int OUTBF, int FULL>
__global__ __launch_bounds__(256) void gemm_k(
    const __bf16* __restrict__ A, const __bf16* __restrict__ B,
    const float* __restrict__ bias, void* __restrict__ outp,
    int M, int N, int K, int lda, int ldb,
    long sAo, long sAi, long sBo, long sBi, long sOo, long sOi, int bmod,
    long orow, long ocol, float alpha)
{
  constexpr int LDS_A = 48, LDS_B = 48;   // padded rows: 96B stride, 16B aligned
  __shared__ __bf16 sA[128 * LDS_A];
  __shared__ __bf16 sB[128 * LDS_B];

  const int z = blockIdx.z, zo = z / bmod, zi = z % bmod;
  const __bf16* Ab = A + zo * sAo + zi * sAi;
  const __bf16* Bb = B + zo * sBo + zi * sBi;
  const long obase = zo * sOo + zi * sOi;

  const int m0 = blockIdx.y * 128, n0 = blockIdx.x * 128;
  const int tid = threadIdx.x, lane = tid & 31, wid = tid >> 5;
  const int wm = wid & 3, wn = wid >> 2;        // 4 waves over M, 2 over N
  const int lane_lo = lane & 15, sel = lane >> 4;
  const int rr = tid >> 2, qq = tid & 3;        // staging: row rr/rr+64, 16B chunk qq

  v8f acc[2][4];
#pragma unroll
  for (int i = 0; i < 2; i++)
#pragma unroll
    for (int j = 0; j < 4; j++)
#pragma unroll
      for (int r = 0; r < 8; r++) acc[i][j][r] = 0.0f;

  for (int k0 = 0; k0 < K; k0 += 32) {
    if (FULL && TRANSB) {
      // --- async staging: memory -> LDS, no VGPR round trip, no guards ---
      async_cp16(&sA[rr * LDS_A + qq * 8],
                 Ab + (size_t)(m0 + rr) * lda + k0 + qq * 8);
      async_cp16(&sA[(rr + 64) * LDS_A + qq * 8],
                 Ab + (size_t)(m0 + rr + 64) * lda + k0 + qq * 8);
      async_cp16(&sB[rr * LDS_B + qq * 8],
                 Bb + (size_t)(n0 + rr) * ldb + k0 + qq * 8);
      async_cp16(&sB[(rr + 64) * LDS_B + qq * 8],
                 Bb + (size_t)(n0 + rr + 64) * ldb + k0 + qq * 8);
      wait_async0();
    } else {
      // --- stage A tile [128 x 32] (branchless clamp + select) ---
#pragma unroll
      for (int half = 0; half < 2; half++) {
        int r = rr + half * 64, gm = m0 + r;
        int gmc = (FULL || gm < M) ? gm : (M - 1);
        v8bf v = *(const v8bf*)(Ab + (size_t)gmc * lda + k0 + qq * 8);
        if (!FULL && gm >= M) v = zero8();
        *(v8bf*)&sA[r * LDS_A + qq * 8] = v;
      }
      // --- stage B tile as [n][k] ---
      if (TRANSB) {
#pragma unroll
        for (int half = 0; half < 2; half++) {
          int r = rr + half * 64, gn = n0 + r;
          int gnc = (gn < N) ? gn : (N - 1);
          v8bf v = *(const v8bf*)(Bb + (size_t)gnc * ldb + k0 + qq * 8);
          if (gn >= N) v = zero8();
          *(v8bf*)&sB[r * LDS_B + qq * 8] = v;
        }
      } else {
        int n = tid & 127, kh = tid >> 7, gn = n0 + n;
        int gnc = (FULL || gn < N) ? gn : (N - 1);
#pragma unroll
        for (int kk = 0; kk < 16; kk++) {
          int k = kh * 16 + kk;
          __bf16 v = Bb[(size_t)(k0 + k) * ldb + gnc];
          if (!FULL && gn >= N) v = (__bf16)0.0f;
          sB[n * LDS_B + k] = v;
        }
      }
    }
    if (k0 + 32 < K)
      __builtin_prefetch(Ab + (size_t)(m0 + (tid >> 1)) * lda + k0 + 32, 0, 1);
    __syncthreads();

    // --- fragments (match ISA 16-bit A/B lane layouts) ---
    v16bf af[2], bfr[4];
#pragma unroll
    for (int mt = 0; mt < 2; mt++) {
      const __bf16* p = &sA[(wm * 32 + mt * 16 + lane_lo) * LDS_A + sel * 8];
      af[mt] = mk16(*(const v8bf*)p, *(const v8bf*)(p + 16));
    }
#pragma unroll
    for (int nt = 0; nt < 4; nt++) {
      const __bf16* p = &sB[(wn * 64 + nt * 16 + lane_lo) * LDS_B + sel * 8];
      bfr[nt] = mk16(*(const v8bf*)p, *(const v8bf*)(p + 16));
    }
#pragma unroll
    for (int mt = 0; mt < 2; mt++)
#pragma unroll
      for (int nt = 0; nt < 4; nt++)
        acc[mt][nt] = __builtin_amdgcn_wmma_f32_16x16x32_bf16(
            false, af[mt], false, bfr[nt], (short)0, acc[mt][nt], false, false);
    __syncthreads();
  }

  // --- epilogue (C layout: vgpr r -> row r + sel*8, col lane_lo) ---
#pragma unroll
  for (int mt = 0; mt < 2; mt++)
#pragma unroll
    for (int nt = 0; nt < 4; nt++) {
      int col = n0 + wn * 64 + nt * 16 + lane_lo;
      if (FULL || col < N) {
        float bv = bias ? bias[col] : 0.0f;
#pragma unroll
        for (int r = 0; r < 8; r++) {
          int row = m0 + wm * 32 + mt * 16 + sel * 8 + r;
          if (FULL || row < M) {
            float v = acc[mt][nt][r] * alpha + bv;
            if (ACT == 1) v = fmaxf(v, 0.0f);
            else if (ACT == 2) v = 1.0f / (1.0f + __expf(-v));
            long off = obase + (long)row * orow + (long)col * ocol;
            if (OUTBF) ((__bf16*)outp)[off] = (__bf16)v;
            else       ((float*)outp)[off]  = v;
          }
        }
      }
    }
}

// ---------------------------------------------------------------------------
// LayerNorm over last dim D (f32 in, f32/bf16 out, in-place safe)
// ---------------------------------------------------------------------------
template <typename OutT>
__global__ __launch_bounds__(128) void ln_k(const float* __restrict__ x,
    const float* __restrict__ w, const float* __restrict__ b,
    OutT* __restrict__ out, int D)
{
  long row = blockIdx.x;
  const float* xr = x + row * D;
  OutT* orow = out + row * D;
  float s = 0.0f, s2 = 0.0f;
  for (int i = threadIdx.x; i < D; i += 128) { float v = xr[i]; s += v; s2 += v * v; }
  for (int m = 16; m; m >>= 1) { s += __shfl_xor(s, m); s2 += __shfl_xor(s2, m); }
  __shared__ float rs[4], rs2[4];
  int wid = threadIdx.x >> 5, lane = threadIdx.x & 31;
  if (lane == 0) { rs[wid] = s; rs2[wid] = s2; }
  __syncthreads();
  s = rs[0] + rs[1] + rs[2] + rs[3];
  s2 = rs2[0] + rs2[1] + rs2[2] + rs2[3];
  float mu = s / D, var = s2 / D - mu * mu;
  float inv = rsqrtf(var + 1e-5f);
  for (int i = threadIdx.x; i < D; i += 128) {
    float v = (xr[i] - mu) * inv * w[i] + b[i];
    orow[i] = (OutT)v;
  }
}

// ---------------------------------------------------------------------------
// Softmax over 256 cols, one wave per row, with bias-broadcast add.
// mode 0 (row attn, H=8): row = h*256+i, bias[(i*256+j)*8+h]
// mode 1 (tri attn, H=4): row = (dl*4+h)*256+i, bias[((l0+dl)*256+j)*4+h]
// ---------------------------------------------------------------------------
__global__ __launch_bounds__(256) void softmax_k(const float* __restrict__ scores,
    const float* __restrict__ bias, __bf16* __restrict__ Pout, int mode, int l0)
{
  int row = blockIdx.x * 8 + (threadIdx.x >> 5);
  int lane = threadIdx.x & 31;
  int Hn; long bbase;
  if (mode == 0) { Hn = 8; int h = row >> 8; int i = row & 255; bbase = (long)i * 2048 + h; }
  else { Hn = 4; int dl = row >> 10; int h = (row >> 8) & 3; bbase = ((long)(l0 + dl) * 256) * 4 + h; }
  const float* sr = scores + (long)row * 256;
  float v[8]; float mx = -1e30f;
#pragma unroll
  for (int t = 0; t < 8; t++) {
    int j = lane + t * 32;
    float x = sr[j] + bias[bbase + (long)j * Hn];
    v[t] = x; mx = fmaxf(mx, x);
  }
  for (int m = 16; m; m >>= 1) mx = fmaxf(mx, __shfl_xor(mx, m));
  float s = 0.0f;
#pragma unroll
  for (int t = 0; t < 8; t++) { v[t] = __expf(v[t] - mx); s += v[t]; }
  for (int m = 16; m; m >>= 1) s += __shfl_xor(s, m);
  float inv = 1.0f / s;
  __bf16* pr = Pout + (long)row * 256;
#pragma unroll
  for (int t = 0; t < 8; t++) pr[lane + t * 32] = (__bf16)(v[t] * inv);
}

// ---------------------------------------------------------------------------
// Elementwise helpers
// ---------------------------------------------------------------------------
__global__ void wconv_k(const float* __restrict__ w, __bf16* __restrict__ wt, int K, int N) {
  long idx = (long)blockIdx.x * 256 + threadIdx.x;
  if (idx >= (long)K * N) return;
  int n = (int)(idx / K), k = (int)(idx % K);
  wt[idx] = (__bf16)w[(long)k * N + n];            // wt[N][K]
}
__global__ void cvt_k(__bf16* __restrict__ d, const float* __restrict__ s, long n) {
  long i = (long)blockIdx.x * 256 + threadIdx.x; if (i < n) d[i] = (__bf16)s[i];
}
__global__ void add_k(float* __restrict__ d, const float* __restrict__ a,
                      const float* __restrict__ b, long n) {
  long i = (long)blockIdx.x * 256 + threadIdx.x; if (i < n) d[i] = a[i] + b[i];
}
__global__ void fmaadd_k(float* __restrict__ d, const float* __restrict__ g,
                         const float* __restrict__ o, long n) {
  long i = (long)blockIdx.x * 256 + threadIdx.x; if (i < n) d[i] += g[i] * o[i];
}
__global__ void gmul_k(__bf16* __restrict__ d, const float* __restrict__ g,
                       const float* __restrict__ o, long n) {
  long i = (long)blockIdx.x * 256 + threadIdx.x; if (i < n) d[i] = (__bf16)(g[i] * o[i]);
}
// outer[i,j,d] = a[i,d]*b[j,d]  (n = 256*256*128)
__global__ void outer_k(__bf16* __restrict__ d, const float* __restrict__ a,
                        const float* __restrict__ b) {
  long idx = (long)blockIdx.x * 256 + threadIdx.x;
  if (idx >= (long)256 * 256 * 128) return;
  int dd = (int)(idx & 127); long t = idx >> 7;
  int j = (int)(t & 255), i = (int)(t >> 8);
  d[idx] = (__bf16)(a[i * 128 + dd] * b[j * 128 + dd]);
}
// d-major scatter: src[(i1*256+i2)*128+d] -> dst[d][r][c], (r,c)=(i1,i2) or swap
__global__ void muldm_k(__bf16* __restrict__ dst, const float* __restrict__ g,
                        const float* __restrict__ v, int swap) {
  long idx = (long)blockIdx.x * 256 + threadIdx.x;
  if (idx >= (long)256 * 256 * 128) return;
  int dd = (int)(idx & 127); long t = idx >> 7;
  int i2 = (int)(t & 255), i1 = (int)(t >> 8);
  int r = swap ? i2 : i1, c = swap ? i1 : i2;
  dst[(long)dd * 65536 + r * 256 + c] = (__bf16)(g[idx] * v[idx]);
}
__global__ void transpose_k(float* __restrict__ dst, const float* __restrict__ src) {
  long idx = (long)blockIdx.x * 256 + threadIdx.x;
  if (idx >= (long)256 * 256 * 128) return;
  int dd = (int)(idx & 127); long t = idx >> 7;
  int j = (int)(t & 255), i = (int)(t >> 8);
  dst[((long)j * 256 + i) * 128 + dd] = src[idx];
}
__global__ void addt_k(float* __restrict__ pair, const float* __restrict__ delta) {
  long idx = (long)blockIdx.x * 256 + threadIdx.x;
  if (idx >= (long)256 * 256 * 128) return;
  int dd = (int)(idx & 127); long t = idx >> 7;
  int j = (int)(t & 255), i = (int)(t >> 8);
  pair[idx] += delta[((long)j * 256 + i) * 128 + dd];
}

// ---------------------------------------------------------------------------
// Host-side GEMM dispatch
// ---------------------------------------------------------------------------
static void gemm_launch(hipStream_t st, int TB, int ACT, int OBF,
    const __bf16* A, const __bf16* B, const float* bias, void* out,
    int M, int N, int K, int lda, int ldb,
    long sAo, long sAi, long sBo, long sBi, long sOo, long sOi, int bmod, int batch,
    long orow, long ocol, float alpha)
{
  dim3 grid((unsigned)((N + 127) / 128), (unsigned)((M + 127) / 128), (unsigned)batch);
  dim3 blk(256);
  const int FULL = (M % 128 == 0) && (N % 128 == 0);
#define GCASE(tb, act, obf, fl)                                                 \
  if (TB == tb && ACT == act && OBF == obf && FULL == fl) {                     \
    gemm_k<tb, act, obf, fl><<<grid, blk, 0, st>>>(A, B, bias, out, M, N, K,    \
        lda, ldb, sAo, sAi, sBo, sBi, sOo, sOi, bmod, orow, ocol, alpha);       \
    return;                                                                     \
  }
  GCASE(1, 0, 0, 1) GCASE(1, 0, 1, 1) GCASE(1, 1, 1, 1) GCASE(1, 2, 0, 1) GCASE(0, 0, 0, 1)
  GCASE(1, 0, 0, 0) GCASE(1, 0, 1, 0) GCASE(1, 1, 1, 0) GCASE(1, 2, 0, 0) GCASE(0, 0, 0, 0)
#undef GCASE
}
static void gemm1(hipStream_t st, int TB, int ACT, int OBF,
    const __bf16* A, const __bf16* B, const float* bias, void* out,
    int M, int N, int K, int lda, int ldb, long orow, float alpha) {
  gemm_launch(st, TB, ACT, OBF, A, B, bias, out, M, N, K, lda, ldb,
              0, 0, 0, 0, 0, 0, 1, 1, orow, 1, alpha);
}

static inline int eg(long n) { return (int)((n + 255) / 256); }

// weights needing transpose+bf16: {input index, K, N}
static const int wlist[][3] = {
  {5,256,256},{6,256,256},{7,256,256},{8,256,256},{10,256,256},{12,128,8},
  {15,256,1024},{17,1024,256},{21,256,128},{23,256,128},{25,128,128},
  {31,128,128},{33,128,128},{35,128,128},{37,128,128},{41,128,128},{43,128,128},
  {47,128,128},{49,128,128},{51,128,128},{53,128,128},{57,128,128},{59,128,128},
  {63,128,128},{64,128,128},{65,128,128},{66,128,128},{68,128,128},{70,128,4},
  {73,128,128},{74,128,128},{75,128,128},{76,128,128},{78,128,128},{80,128,4},
  {83,128,512},{85,512,128}};

extern "C" void kernel_launch(void* const* d_in, const int* in_sizes, int n_in,
                              void* d_out, int out_size, void* d_ws, size_t ws_size,
                              hipStream_t stream) {
  (void)in_sizes; (void)n_in; (void)out_size; (void)ws_size;
  const int L = 256, LL = 65536, DP = 128, DN = 256;
  const float SCALE = 0.17677669529663687f;  // 1/sqrt(32)

  char* base = (char*)d_ws; size_t off = 0;
  auto alloc = [&](size_t bytes) -> void* {
    void* p = base + off; off = (off + bytes + 255) & ~(size_t)255; return p;
  };
  auto prm = [&](int i) { return (const float*)d_in[i]; };

  // --- transpose/convert weights to bf16 [N][K] ---
  __bf16* wt[96] = {nullptr};
  for (auto& e : wlist) {
    int idx = e[0], K = e[1], N = e[2];
    wt[idx] = (__bf16*)alloc((size_t)K * N * 2);
    wconv_k<<<eg((long)K * N), 256, 0, stream>>>(prm(idx), wt[idx], K, N);
  }

  // --- activations / scratch ---
  float*  single_cur = (float*)alloc(65536 * 4);
  __bf16* xs   = (__bf16*)alloc(65536 * 2);
  __bf16* qrow = (__bf16*)alloc(65536 * 2);
  __bf16* krow = (__bf16*)alloc(65536 * 2);
  __bf16* vrow = (__bf16*)alloc(65536 * 2);
  float*  grow = (float*)alloc(65536 * 4);
  float*  obuf_s = (float*)alloc(65536 * 4);
  __bf16* ob   = (__bf16*)alloc(65536 * 2);
  __bf16* sh   = (__bf16*)alloc((size_t)256 * 1024 * 2);
  float*  s_tmp = (float*)alloc((size_t)256 * 1024 * 4);
  float*  av_s = (float*)alloc((size_t)256 * 128 * 4);
  float*  bv_s = (float*)alloc((size_t)256 * 128 * 4);

  size_t PAIRB = (size_t)LL * DP;
  float*  pair_cur = (float*)alloc(PAIRB * 4);
  float*  pairT    = (float*)alloc(PAIRB * 4);
  float*  buf1 = (float*)alloc(PAIRB * 4);
  float*  buf2 = (float*)alloc(PAIRB * 4);
  float*  buf3 = (float*)alloc(PAIRB * 4);
  __bf16* xn   = (__bf16*)alloc(PAIRB * 2);
  __bf16* bfA  = (__bf16*)alloc(PAIRB * 2);
  __bf16* bfB  = (__bf16*)alloc(PAIRB * 2);
  __bf16* bfC  = (__bf16*)alloc(PAIRB * 2);
  __bf16* bfD  = (__bf16*)alloc(PAIRB * 2);
  float*  bias_row = (float*)alloc((size_t)LL * 8 * 4);
  float*  bias_tri = (float*)alloc((size_t)LL * 4 * 4);
  float*  scores = (float*)alloc((size_t)64 * 65536 * 4);
  __bf16* Pbuf   = (__bf16*)alloc((size_t)64 * 65536 * 2);

  float* out_single = (float*)d_out;
  float* out_pair   = out_single + 65536;

  hipMemcpyAsync(single_cur, d_in[0], 65536 * 4, hipMemcpyDeviceToDevice, stream);
  hipMemcpyAsync(pair_cur, d_in[1], PAIRB * 4, hipMemcpyDeviceToDevice, stream);

  // ===================== Phase 1: row attention ==========================
  ln_k<__bf16><<<L, 128, 0, stream>>>(single_cur, prm(3), prm(4), xs, DN);
  gemm1(stream, 1, 0, 1, xs, wt[5], nullptr, qrow, L, DN, DN, DN, DN, DN, 1.0f);
  gemm1(stream, 1, 0, 1, xs, wt[6], nullptr, krow, L, DN, DN, DN, DN, DN, 1.0f);
  gemm1(stream, 1, 0, 1, xs, wt[7], nullptr, vrow, L, DN, DN, DN, DN, DN, 1.0f);
  gemm1(stream, 1, 2, 0, xs, wt[8], prm(9), grow, L, DN, DN, DN, DN, DN, 1.0f);
  cvt_k<<<eg(PAIRB), 256, 0, stream>>>(bfD, pair_cur, (long)PAIRB);
  gemm1(stream, 1, 0, 0, bfD, wt[12], nullptr, bias_row, LL, 8, DP, DP, DP, 8, 1.0f);
  // scores[h][i][j] = scale * q_h k_h^T   (batch 8 heads)
  gemm_launch(stream, 1, 0, 0, qrow, krow, nullptr, scores, L, L, 32, DN, DN,
              32, 0, 32, 0, 65536, 0, 1, 8, 256, 1, SCALE);
  softmax_k<<<(8 * 256) / 8, 256, 0, stream>>>(scores, bias_row, Pbuf, 0, 0);
  // O[h] = P_h V_h  -> obuf_s[i][h*32+n]
  gemm_launch(stream, 0, 0, 0, Pbuf, vrow, nullptr, obuf_s, L, 32, L, L, DN,
              65536, 0, 32, 0, 32, 0, 1, 8, 256, 1, 1.0f);
  gmul_k<<<eg(65536), 256, 0, stream>>>(ob, grow, obuf_s, 65536);
  gemm1(stream, 1, 0, 0, ob, wt[10], prm(11), s_tmp, L, DN, DN, DN, DN, DN, 1.0f);
  add_k<<<eg(65536), 256, 0, stream>>>(single_cur, single_cur, s_tmp, 65536);

  // ===================== Phase 2: single FFN =============================
  ln_k<__bf16><<<L, 128, 0, stream>>>(single_cur, prm(13), prm(14), xs, DN);
  gemm1(stream, 1, 1, 1, xs, wt[15], prm(16), sh, L, 1024, DN, DN, DN, 1024, 1.0f);
  gemm1(stream, 1, 0, 0, sh, wt[17], prm(18), s_tmp, L, DN, 1024, 1024, 1024, DN, 1.0f);
  add_k<<<eg(65536), 256, 0, stream>>>(single_cur, single_cur, s_tmp, 65536);

  // ===================== Phase 3: pair update ============================
  ln_k<__bf16><<<L, 128, 0, stream>>>(single_cur, prm(19), prm(20), xs, DN);
  gemm1(stream, 1, 0, 0, xs, wt[21], prm(22), av_s, L, DP, DN, DN, DN, DP, 1.0f);
  gemm1(stream, 1, 0, 0, xs, wt[23], prm(24), bv_s, L, DP, DN, DN, DN, DP, 1.0f);
  outer_k<<<eg(PAIRB), 256, 0, stream>>>(bfD, av_s, bv_s);
  gemm1(stream, 1, 0, 0, bfD, wt[25], prm(26), buf1, LL, DP, DP, DP, DP, DP, 1.0f);
  add_k<<<eg(PAIRB), 256, 0, stream>>>(pair_cur, pair_cur, buf1, (long)PAIRB);
  ln_k<float><<<LL, 128, 0, stream>>>(pair_cur, prm(27), prm(28), pair_cur, DP);

  // ===================== Phase 4: triangle multiplications ===============
  auto tri_mult = [&](int p0, int swap) {
    ln_k<__bf16><<<LL, 128, 0, stream>>>(pair_cur, prm(p0), prm(p0 + 1), xn, DP);
    gemm1(stream, 1, 2, 0, xn, wt[p0 + 6], prm(p0 + 7), buf2, LL, DP, DP, DP, DP, DP, 1.0f); // sig(gate_a)
    gemm1(stream, 1, 0, 0, xn, wt[p0 + 2], prm(p0 + 3), buf1, LL, DP, DP, DP, DP, DP, 1.0f); // a
    muldm_k<<<eg(PAIRB), 256, 0, stream>>>(bfA, buf2, buf1, swap);
    gemm1(stream, 1, 2, 0, xn, wt[p0 + 8], prm(p0 + 9), buf2, LL, DP, DP, DP, DP, DP, 1.0f); // sig(gate_b)
    gemm1(stream, 1, 0, 0, xn, wt[p0 + 4], prm(p0 + 5), buf1, LL, DP, DP, DP, DP, DP, 1.0f); // b
    muldm_k<<<eg(PAIRB), 256, 0, stream>>>(bfB, buf2, buf1, swap);
    gemm1(stream, 1, 2, 0, xn, wt[p0 + 14], prm(p0 + 15), buf2, LL, DP, DP, DP, DP, DP, 1.0f); // sig(g)
    // pp[i,j,d]: d-batched 256x256x256 NT GEMM, strided scatter into [i,j,d]
    gemm_launch(stream, 1, 0, 0, bfA, bfB, nullptr, buf1, 256, 256, 256, 256, 256,
                65536, 0, 65536, 0, 1, 0, 1, 128, 32768, 128, 1.0f);
    ln_k<__bf16><<<LL, 128, 0, stream>>>(buf1, prm(p0 + 10), prm(p0 + 11), xn, DP);
    gemm1(stream, 1, 0, 0, xn, wt[p0 + 12], prm(p0 + 13), buf3, LL, DP, DP, DP, DP, DP, 1.0f);
    fmaadd_k<<<eg(PAIRB), 256, 0, stream>>>(pair_cur, buf2, buf3, (long)PAIRB);
  };
  tri_mult(29, 0);  // outgoing
  tri_mult(45, 1);  // incoming

  // ===================== Phase 5: triangle attention ======================
  auto tri_attn = [&](int p0, int end) {
    const float* PT = pair_cur;
    if (end) {
      transpose_k<<<eg(PAIRB), 256, 0, stream>>>(pairT, pair_cur);
      PT = pairT;
    }
    ln_k<__bf16><<<LL, 128, 0, stream>>>(PT, prm(p0), prm(p0 + 1), xn, DP);
    gemm1(stream, 1, 0, 1, xn, wt[p0 + 2], nullptr, bfA, LL, DP, DP, DP, DP, DP, 1.0f); // q
    gemm1(stream, 1, 0, 1, xn, wt[p0 + 3], nullptr, bfB, LL, DP, DP, DP, DP, DP, 1.0f); // k
    gemm1(stream, 1, 0, 1, xn, wt[p0 + 4], nullptr, bfC, LL, DP, DP, DP, DP, DP, 1.0f); // v
    gemm1(stream, 1, 2, 0, xn, wt[p0 + 5], prm(p0 + 6), buf2, LL, DP, DP, DP, DP, DP, 1.0f); // sig(g)
    cvt_k<<<eg(PAIRB), 256, 0, stream>>>(bfD, PT, (long)PAIRB);   // bias from pre-norm pair
    gemm1(stream, 1, 0, 0, bfD, wt[p0 + 9], nullptr, bias_tri, LL, 4, DP, DP, DP, 4, 1.0f);
    for (int l0 = 0; l0 < 256; l0 += 16) {
      long boff = (long)l0 * 32768;
      // scores[dl*4+h][i][j], batch 64 = 16 l-rows x 4 heads
      gemm_launch(stream, 1, 0, 0, bfA + boff, bfB + boff, nullptr, scores,
                  256, 256, 32, DP, DP, 32768, 32, 32768, 32, 262144, 65536,
                  4, 64, 256, 1, SCALE);
      softmax_k<<<(64 * 256) / 8, 256, 0, stream>>>(scores, bias_tri, Pbuf, 1, l0);
      gemm_launch(stream, 0, 0, 0, Pbuf, bfC + boff, nullptr, buf3 + boff,
                  256, 32, 256, 256, DP, 262144, 65536, 32768, 32, 32768, 32,
                  4, 64, DP, 1, 1.0f);
    }
    gmul_k<<<eg(PAIRB), 256, 0, stream>>>(bfD, buf2, buf3, (long)PAIRB);
    gemm1(stream, 1, 0, 0, bfD, wt[p0 + 7], prm(p0 + 8), buf3, LL, DP, DP, DP, DP, DP, 1.0f);
    if (end) addt_k<<<eg(PAIRB), 256, 0, stream>>>(pair_cur, buf3);
    else     add_k<<<eg(PAIRB), 256, 0, stream>>>(pair_cur, pair_cur, buf3, (long)PAIRB);
  };
  tri_attn(61, 0);  // start
  tri_attn(71, 1);  // end

  // ===================== Phase 6: pair FFN (chunked) ======================
  ln_k<__bf16><<<LL, 128, 0, stream>>>(pair_cur, prm(81), prm(82), xn, DP);
  for (int c = 0; c < 4; c++) {
    long r0 = (long)c * 16384;
    gemm1(stream, 1, 1, 1, xn + r0 * DP, wt[83], prm(84), bfA,
          16384, 512, DP, DP, DP, 512, 1.0f);
    gemm1(stream, 1, 0, 0, bfA, wt[85], prm(86), buf1 + r0 * DP,
          16384, DP, 512, 512, 512, DP, 1.0f);
  }
  add_k<<<eg(PAIRB), 256, 0, stream>>>(out_pair, pair_cur, buf1, (long)PAIRB);
  hipMemcpyAsync(out_single, single_cur, 65536 * 4, hipMemcpyDeviceToDevice, stream);
}
